// Attention_56470230008033
// MI455X (gfx1250) — matrix operand order repaired and checked by reading, not compiled, not run on hardware
//
#include <hip/hip_runtime.h>
#include <hip/hip_bf16.h>

// ---------------------------------------------------------------------------
// Fused attention forward for MI455X (gfx1250, wave32, WMMA + async-to-LDS).
// Stage 1: QKV GEMM  (fp32 in -> bf16 Q/K/V in [B][H][N][D] workspace)
// Stage 2: flash attention (bf16 WMMA QK^T and P*V, fp32 online softmax,
//          Q/K tiles streamed with GLOBAL_LOAD_ASYNC_TO_LDS_B128)
// Stage 3: output projection (async bf16 A tiles x bf16(W_proj) -> fp32 out)
// ---------------------------------------------------------------------------

typedef __bf16 bf16;
typedef __attribute__((ext_vector_type(16))) __bf16 v16bf;
typedef __attribute__((ext_vector_type(8)))  __bf16 v8bf;
typedef __attribute__((ext_vector_type(4)))  __bf16 v4bf;
typedef __attribute__((ext_vector_type(8)))  float  v8f;
typedef __attribute__((ext_vector_type(4)))  float  v4f;

#define BDIM  2
#define SEQ   2048
#define CDIM  1024
#define HEADS 16
#define HD    64

__device__ __forceinline__ v8f zero8() {
  v8f z;
#pragma unroll
  for (int i = 0; i < 8; ++i) z[i] = 0.0f;
  return z;
}

// Async global->LDS 16-byte copy (ASYNCcnt-tracked, bypasses VGPRs).
// LDS byte offset is the low 32 bits of the flat shared-memory pointer.
__device__ __forceinline__ void async_load_b128(void* lds_dst, const void* gsrc) {
  const unsigned lds = (unsigned)(uintptr_t)lds_dst;
  asm volatile("global_load_async_to_lds_b128 %0, %1, off"
               :: "v"(lds), "v"(gsrc) : "memory");
}
__device__ __forceinline__ void wait_async0() {
  asm volatile("s_wait_asynccnt 0" ::: "memory");
}

// A-fragment (16x32 bf16) from row-major LDS tile [16][ld], ld multiple of 8.
// Lane L: row M=L&15, K = h*8 + 0..7 and 16 + h*8 + 0..7  (h = L>=16).
__device__ __forceinline__ v16bf load_frag_a(const bf16* base, int ld) {
  const int lane = threadIdx.x & 31;
  const int row  = lane & 15;
  const int hh   = lane >> 4;
  const bf16* p = base + row * ld + hh * 8;
  v8bf lo = *(const v8bf*)(p);
  v8bf hi = *(const v8bf*)(p + 16);
  v16bf f;
#pragma unroll
  for (int j = 0; j < 8; ++j) { f[j] = lo[j]; f[j + 8] = hi[j]; }
  return f;
}

// B-fragment (32x16 bf16) from LDS tile stored transposed [n][k], ld mult of 16.
// Lane L: col N = L&15, K = h*8 + 0..7 and 16 + h*8 + 0..7  (h = L>=16).
__device__ __forceinline__ v16bf load_frag_b(const bf16* baseT, int ld) {
  const int lane = threadIdx.x & 31;
  const int col  = lane & 15;
  const int hh   = lane >> 4;
  const bf16* p = baseT + col * ld + hh * 8;
  v8bf lo = *(const v8bf*)(p);
  v8bf hi = *(const v8bf*)(p + 16);
  v16bf f;
#pragma unroll
  for (int j = 0; j < 8; ++j) { f[j] = lo[j]; f[j + 8] = hi[j]; }
  return f;
}

__device__ __forceinline__ v8f wmma_bf16(v16bf a, v16bf b, v8f c) {
  return __builtin_amdgcn_wmma_f32_16x16x32_bf16(false, a, false, b,
                                                 (short)0, c, false, false);
}

// ---------------------------------------------------------------------------
// Stage 1: QKV GEMM.  X[4096][1024] fp32 x W[1024][3072] fp32 + bias,
// scattered as bf16 Q/K/V in [B][H][N][D].  (fp32 sources need conversion, so
// this stage keeps the load-convert-store LDS path.)
// ---------------------------------------------------------------------------
__global__ __launch_bounds__(256) void qkv_gemm_kernel(
    const float* __restrict__ X, const float* __restrict__ W,
    const float* __restrict__ bias,
    bf16* __restrict__ Qw, bf16* __restrict__ Kw, bf16* __restrict__ Vw) {
  constexpr int BM = 128, BN = 128, BK = 32;
  constexpr int LDA = BK + 8;    // 40
  constexpr int LDB = BK + 16;   // 48
  __shared__ bf16 As [BM * LDA];
  __shared__ bf16 BsT[BN * LDB];

  const int tid  = threadIdx.x;
  const int wave = tid >> 5;
  const int lane = tid & 31;
  const int gm0  = blockIdx.y * BM;
  const int gn0  = blockIdx.x * BN;
  const int wm   = (wave >> 2) * 64;
  const int wn   = (wave & 3) * 32;

  v8f acc[4][2];
#pragma unroll
  for (int mt = 0; mt < 4; ++mt)
#pragma unroll
    for (int nt = 0; nt < 2; ++nt) acc[mt][nt] = zero8();

  for (int k0 = 0; k0 < CDIM; k0 += BK) {
    __syncthreads();
    {  // A tile: fp32 -> bf16
      const int r0 = tid >> 3;
      const int kk = (tid & 7) * 4;
#pragma unroll
      for (int i = 0; i < 4; ++i) {
        const int r = r0 + i * 32;
        v4f x = *(const v4f*)(X + (size_t)(gm0 + r) * CDIM + k0 + kk);
        v4bf xb; xb[0]=(bf16)x[0]; xb[1]=(bf16)x[1]; xb[2]=(bf16)x[2]; xb[3]=(bf16)x[3];
        *(v4bf*)(As + r * LDA + kk) = xb;
      }
    }
    {  // B tile: fp32 -> bf16, stored transposed [n][k]
      const int kr = tid >> 5;
      const int nn = (tid & 31) * 4;
#pragma unroll
      for (int i = 0; i < 4; ++i) {
        const int k = kr + i * 8;
        v4f w = *(const v4f*)(W + (size_t)(k0 + k) * (3 * CDIM) + gn0 + nn);
#pragma unroll
        for (int j = 0; j < 4; ++j) BsT[(nn + j) * LDB + k] = (bf16)w[j];
      }
    }
    if (k0 + BK < CDIM) {  // prefetch next tiles -> global_prefetch_b8
      __builtin_prefetch(X + (size_t)(gm0 + (tid >> 3)) * CDIM + k0 + BK, 0, 3);
      __builtin_prefetch(W + (size_t)(k0 + BK + (tid >> 5)) * (3 * CDIM) + gn0, 0, 3);
    }
    __syncthreads();

    v16bf bfrag[2];
#pragma unroll
    for (int nt = 0; nt < 2; ++nt)
      bfrag[nt] = load_frag_b(BsT + (wn + nt * 16) * LDB, LDB);
#pragma unroll
    for (int mt = 0; mt < 4; ++mt) {
      v16bf a = load_frag_a(As + (wm + mt * 16) * LDA, LDA);
#pragma unroll
      for (int nt = 0; nt < 2; ++nt)
        acc[mt][nt] = wmma_bf16(a, bfrag[nt], acc[mt][nt]);
    }
  }

  // Epilogue: bias + scatter into [B][H][N][D] bf16 Q/K/V
  const int colL = lane & 15;
  const int rofs = (lane >> 4) * 8;
#pragma unroll
  for (int mt = 0; mt < 4; ++mt)
#pragma unroll
    for (int nt = 0; nt < 2; ++nt)
#pragma unroll
      for (int i = 0; i < 8; ++i) {
        const int grow = gm0 + wm + mt * 16 + i + rofs;
        const int gcol = gn0 + wn + nt * 16 + colL;
        const float v = acc[mt][nt][i] + bias[gcol];
        const int sec = gcol >> 10;          // 0:Q 1:K 2:V
        const int c   = gcol & 1023;
        const int h   = c >> 6;
        const int d   = c & 63;
        const int b   = grow >> 11;
        const int r   = grow & 2047;
        bf16* dst = (sec == 0) ? Qw : (sec == 1) ? Kw : Vw;
        dst[(((size_t)(b * HEADS + h)) * SEQ + r) * HD + d] = (bf16)v;
      }
}

// ---------------------------------------------------------------------------
// Stage 2: flash attention. One block per (b,h, 128-row Q tile); 8 waves each
// own 16 Q rows. Q and K are pure bf16 copies -> async global->LDS DMA;
// V needs a transpose so it goes through VGPRs.
// ---------------------------------------------------------------------------
__global__ __launch_bounds__(256) void attn_kernel(
    const bf16* __restrict__ Qw, const bf16* __restrict__ Kw,
    const bf16* __restrict__ Vw, bf16* __restrict__ Aout) {
  constexpr int BR = 128, BC = 64;
  constexpr int LDQ = HD + 8;    // 72
  constexpr int LDK = HD + 16;   // 80
  constexpr int LDV = BC + 16;   // 80
  constexpr int LDP = BC + 8;    // 72
  __shared__ bf16 Qs[BR * LDQ];        // 18 KB
  __shared__ bf16 Ks[BC * LDK];        // 10 KB   [key][d]
  __shared__ bf16 Vt[HD * LDV];        // 10 KB   [d][key] (transposed)
  __shared__ bf16 Ps[8 * 16 * LDP];    // 18 KB   per-wave P tiles

  const int tid  = threadIdx.x;
  const int wave = tid >> 5;
  const int lane = tid & 31;
  const int bh   = blockIdx.y;
  const int q0   = blockIdx.x * BR;
  const size_t head_base = (size_t)bh * SEQ * HD;
  const bf16* Qg = Qw + head_base + (size_t)q0 * HD;
  const bf16* Kg = Kw + head_base;
  const bf16* Vg = Vw + head_base;

  {  // Q tile: async DMA copy
    const int r0 = tid >> 3;
    const int dg = (tid & 7) * 8;
#pragma unroll
    for (int i = 0; i < 4; ++i) {
      const int r = r0 + i * 32;
      async_load_b128(Qs + r * LDQ + dg, Qg + (size_t)r * HD + dg);
    }
  }

  v8f   O[4];
  float m[8], l[8];
#pragma unroll
  for (int t = 0; t < 4; ++t) O[t] = zero8();
#pragma unroll
  for (int i = 0; i < 8; ++i) { m[i] = -1e30f; l[i] = 0.0f; }

  const bf16* Qbase = Qs + (wave * 16) * LDQ;
  bf16* Pw = Ps + wave * 16 * LDP;
  const int colL = lane & 15;
  const int hh   = lane >> 4;
  constexpr float SCALE = 0.125f;                 // 1/sqrt(64)
  constexpr float LOG2E = 1.4426950408889634f;

  for (int kv = 0; kv < SEQ; kv += BC) {
    __syncthreads();
    {  // K chunk [key][d]: async DMA copy
      const int r0 = tid >> 3;
      const int dg = (tid & 7) * 8;
#pragma unroll
      for (int i = 0; i < 2; ++i) {
        const int key = r0 + i * 32;
        async_load_b128(Ks + key * LDK + dg,
                        Kg + (size_t)(kv + key) * HD + dg);
      }
    }
    {  // V chunk transposed -> [d][key] (register transpose)
      const int r0 = tid >> 3;
      const int dg = (tid & 7) * 8;
#pragma unroll
      for (int i = 0; i < 2; ++i) {
        const int key = r0 + i * 32;
        v8bf vv = *(const v8bf*)(Vg + (size_t)(kv + key) * HD + dg);
#pragma unroll
        for (int j = 0; j < 8; ++j) Vt[(dg + j) * LDV + key] = vv[j];
      }
    }
    wait_async0();
    __syncthreads();

    // S = Q K^T  (4 key tiles x 2 k-steps over D=64)
    v8f S[4];
#pragma unroll
    for (int t = 0; t < 4; ++t) {
      v8f s = zero8();
#pragma unroll
      for (int ks = 0; ks < 2; ++ks) {
        v16bf a = load_frag_a(Qbase + ks * 32, LDQ);
        v16bf b = load_frag_b(Ks + (t * 16) * LDK + ks * 32, LDK);
        s = wmma_bf16(a, b, s);
      }
      S[t] = s;
    }

    // Online softmax: row stats live per half-wave (rows i and i+8).
    float scl[8];
#pragma unroll
    for (int i = 0; i < 8; ++i) {
      float mx = S[0][i];
#pragma unroll
      for (int t = 1; t < 4; ++t) mx = fmaxf(mx, S[t][i]);
#pragma unroll
      for (int off = 1; off < 16; off <<= 1)
        mx = fmaxf(mx, __shfl_xor(mx, off, 32));
      const float mn = fmaxf(m[i], mx * SCALE);
      scl[i] = exp2f((m[i] - mn) * LOG2E);
      m[i] = mn;
    }
    float rs[8];
#pragma unroll
    for (int i = 0; i < 8; ++i) rs[i] = 0.0f;
#pragma unroll
    for (int t = 0; t < 4; ++t)
#pragma unroll
      for (int i = 0; i < 8; ++i) {
        const float p = exp2f((S[t][i] * SCALE - m[i]) * LOG2E);
        rs[i] += p;
        Pw[(i + hh * 8) * LDP + t * 16 + colL] = (bf16)p;
      }
#pragma unroll
    for (int i = 0; i < 8; ++i) {
      float r = rs[i];
#pragma unroll
      for (int off = 1; off < 16; off <<= 1) r += __shfl_xor(r, off, 32);
      l[i] = l[i] * scl[i] + r;
    }
#pragma unroll
    for (int t = 0; t < 4; ++t)
#pragma unroll
      for (int i = 0; i < 8; ++i) O[t][i] *= scl[i];

    // O += P V   (P: [16][64] per-wave LDS, V^T: [d][key])
#pragma unroll
    for (int dt = 0; dt < 4; ++dt)
#pragma unroll
      for (int ks = 0; ks < 2; ++ks) {
        v16bf a = load_frag_a(Pw + ks * 32, LDP);
        v16bf b = load_frag_b(Vt + (dt * 16) * LDV + ks * 32, LDV);
        O[dt] = wmma_bf16(a, b, O[dt]);
      }
  }

  // Normalize and write attn output as bf16 [B][N][C] (C = H*D)
  const int b = bh >> 4;
  const int h = bh & 15;
#pragma unroll
  for (int dt = 0; dt < 4; ++dt)
#pragma unroll
    for (int i = 0; i < 8; ++i) {
      const int r = q0 + wave * 16 + i + hh * 8;
      const int d = h * HD + dt * 16 + colL;
      Aout[((size_t)b * SEQ + r) * CDIM + d] = (bf16)(O[dt][i] / l[i]);
    }
}

// ---------------------------------------------------------------------------
// Stage 3: output projection.  A[4096][1024] bf16 x W_proj[1024][1024] fp32
// -> fp32 out + bias.  A tiles are bf16 copies -> async global->LDS DMA.
// ---------------------------------------------------------------------------
__global__ __launch_bounds__(256) void proj_gemm_kernel(
    const bf16* __restrict__ X, const float* __restrict__ W,
    const float* __restrict__ bias, float* __restrict__ out) {
  constexpr int BM = 128, BN = 128, BK = 32;
  constexpr int LDA = BK + 8;
  constexpr int LDB = BK + 16;
  __shared__ bf16 As [BM * LDA];
  __shared__ bf16 BsT[BN * LDB];

  const int tid  = threadIdx.x;
  const int wave = tid >> 5;
  const int lane = tid & 31;
  const int gm0  = blockIdx.y * BM;
  const int gn0  = blockIdx.x * BN;
  const int wm   = (wave >> 2) * 64;
  const int wn   = (wave & 3) * 32;

  v8f acc[4][2];
#pragma unroll
  for (int mt = 0; mt < 4; ++mt)
#pragma unroll
    for (int nt = 0; nt < 2; ++nt) acc[mt][nt] = zero8();

  for (int k0 = 0; k0 < CDIM; k0 += BK) {
    __syncthreads();
    {  // A tile (already bf16): async DMA copy, 16B per lane per iter
      const int r0 = tid >> 2;            // 0..63
      const int kk = (tid & 3) * 8;       // 0,8,16,24
#pragma unroll
      for (int i = 0; i < 2; ++i) {
        const int r = r0 + i * 64;
        async_load_b128(As + r * LDA + kk,
                        X + (size_t)(gm0 + r) * CDIM + k0 + kk);
      }
    }
    {  // B tile fp32 -> bf16 transposed
      const int kr = tid >> 5;
      const int nn = (tid & 31) * 4;
#pragma unroll
      for (int i = 0; i < 4; ++i) {
        const int k = kr + i * 8;
        v4f w = *(const v4f*)(W + (size_t)(k0 + k) * CDIM + gn0 + nn);
#pragma unroll
        for (int j = 0; j < 4; ++j) BsT[(nn + j) * LDB + k] = (bf16)w[j];
      }
    }
    if (k0 + BK < CDIM) {
      __builtin_prefetch(W + (size_t)(k0 + BK + (tid >> 5)) * CDIM + gn0, 0, 3);
    }
    wait_async0();
    __syncthreads();

    v16bf bfrag[2];
#pragma unroll
    for (int nt = 0; nt < 2; ++nt)
      bfrag[nt] = load_frag_b(BsT + (wn + nt * 16) * LDB, LDB);
#pragma unroll
    for (int mt = 0; mt < 4; ++mt) {
      v16bf a = load_frag_a(As + (wm + mt * 16) * LDA, LDA);
#pragma unroll
      for (int nt = 0; nt < 2; ++nt)
        acc[mt][nt] = wmma_bf16(a, bfrag[nt], acc[mt][nt]);
    }
  }

  const int colL = lane & 15;
  const int rofs = (lane >> 4) * 8;
#pragma unroll
  for (int mt = 0; mt < 4; ++mt)
#pragma unroll
    for (int nt = 0; nt < 2; ++nt)
#pragma unroll
      for (int i = 0; i < 8; ++i) {
        const int grow = gm0 + wm + mt * 16 + i + rofs;
        const int gcol = gn0 + wn + nt * 16 + colL;
        out[(size_t)grow * CDIM + gcol] = acc[mt][nt][i] + bias[gcol];
      }
}

// ---------------------------------------------------------------------------
extern "C" void kernel_launch(void* const* d_in, const int* in_sizes, int n_in,
                              void* d_out, int out_size, void* d_ws,
                              size_t ws_size, hipStream_t stream) {
  (void)in_sizes; (void)n_in; (void)out_size; (void)ws_size;
  const float* q      = (const float*)d_in[0];
  const float* W_qkv  = (const float*)d_in[1];
  const float* b_qkv  = (const float*)d_in[2];
  const float* W_proj = (const float*)d_in[3];
  const float* b_proj = (const float*)d_in[4];
  float* out = (float*)d_out;

  bf16* ws = (bf16*)d_ws;
  const size_t per = (size_t)BDIM * HEADS * SEQ * HD;  // 4M bf16 each
  bf16* Qw = ws;
  bf16* Kw = ws + per;
  bf16* Vw = ws + 2 * per;
  bf16* Aw = ws + 3 * per;

  qkv_gemm_kernel<<<dim3(3 * CDIM / 128, (BDIM * SEQ) / 128), 256, 0, stream>>>(
      q, W_qkv, b_qkv, Qw, Kw, Vw);
  attn_kernel<<<dim3(SEQ / 128, BDIM * HEADS), 256, 0, stream>>>(Qw, Kw, Vw, Aw);
  proj_gemm_kernel<<<dim3(CDIM / 128, (BDIM * SEQ) / 128), 256, 0, stream>>>(
      Aw, W_proj, b_proj, out);
}
